// NPUFusedMLAPreprocess_37623913513199
// MI455X (gfx1250) — compile-verified, matrix operand order checked by reading, not run on hardware
//
#include <hip/hip_runtime.h>
#include <hip/hip_bf16.h>

// MLA preprocess, MI455X (gfx1250, wave32, WMMA).
// GEMMs run as bf16 WMMA (v_wmma_f32_16x16x32_bf16) with f32 accum.
// A-operand staging for bf16-A GEMMs uses GLOBAL_LOAD_ASYNC_TO_LDS_B128
// (ASYNCcnt) when the toolchain exposes the builtins.

#define QLORA  1536
#define KVLORA 512
#define ROPE_D 64
#define NOPE_D 128
#define NHEAD  16
#define QKD    192
#define NTOK   4096
#define HID    7168
#define NSLOT  16384
#define EPS_F  1e-6f

typedef __attribute__((ext_vector_type(16))) __bf16 v16bf;
typedef __attribute__((ext_vector_type(8)))  float  v8f;
typedef __attribute__((ext_vector_type(4)))  int    v4i;

#if defined(__has_builtin)
#if __has_builtin(__builtin_amdgcn_global_load_async_to_lds_b128) && \
    __has_builtin(__builtin_amdgcn_s_wait_asynccnt)
#define USE_ASYNC_LDS 1
#else
#define USE_ASYNC_LDS 0
#endif
#else
#define USE_ASYNC_LDS 0
#endif

__device__ __forceinline__ void async_ld_b128(const void* g, void* l)
{
#if USE_ASYNC_LDS
  // prototype: (v4i addrspace(1)*, v4i addrspace(3)*, imm offset, imm cpol)
  __builtin_amdgcn_global_load_async_to_lds_b128(
      (__attribute__((address_space(1))) v4i*)(g),
      (__attribute__((address_space(3))) v4i*)(l),
      /*offset=*/0, /*cpol=*/0);
#else
  (void)g; (void)l;
#endif
}

__device__ __forceinline__ void wait_async0()
{
#if USE_ASYNC_LDS
  __builtin_amdgcn_s_wait_asynccnt(0);
#endif
}

// ---------------------------------------------------------------------------
// Tiled WMMA GEMM: C[M,N](f32,ldc) = A[M,K](f32 or bf16,lda) * B[K,N](f32,ldb)
// BM=128 BN=64 BK=32; 256 threads = 8 waves laid out 4x2; wave tile 32x32
// (2x2 WMMA 16x16x32 bf16). A/B staged in LDS K-contiguous so each WMMA
// fragment is two ds b128 loads per lane (matches CDNA5 16-bit VGPR layout:
// lanes 0-15 hold K 0-7 & 16-23, lanes 16-31 hold K 8-15 & 24-31).
// When A is bf16 (no cvt needed) it is staged via async global->LDS b128.
// ---------------------------------------------------------------------------
template<bool ABF16>
__global__ __launch_bounds__(256)
void gemm_wmma_bf16(const void* __restrict__ Av, const float* __restrict__ B,
                    float* __restrict__ C, int M, int N, int K,
                    int lda, int ldb, int ldc,
                    long aBatch, long bBatch, long cBatch)
{
  constexpr int BM = 128, BN = 64, BK = 32, LDK = 40;  // LDK pad kills bank conflicts
  constexpr bool ASYNC_A = ABF16 && (USE_ASYNC_LDS != 0);
  __shared__ __align__(16) __bf16 sA[2][BM][LDK];
  __shared__ __align__(16) __bf16 sB[2][BN][LDK];

  const int tid  = threadIdx.x;
  const int lane = tid & 31;
  const int wid  = tid >> 5;
  const int wm   = wid >> 1;            // 0..3
  const int wn   = wid & 1;             // 0..1
  const int bm0  = blockIdx.y * BM;
  const int bn0  = blockIdx.x * BN;
  const int z    = blockIdx.z;

  const float*  Af = (const float*)Av + (ABF16 ? 0 : (long)z * aBatch);
  const __bf16* Ah = (const __bf16*)Av + (ABF16 ? (long)z * aBatch : 0);
  const float*  Bz = B + (long)z * bBatch;
  float*        Cz = C + (long)z * cBatch;

  // loader geometry: A -> 128 rows x 2 half-rows of 16 K-elems
  const int arow = tid >> 1;            // 0..127
  const int aks  = (tid & 1) * 16;      // 0 or 16
  // B loaded transposed: thread owns one column n, 8 consecutive k
  const int bncol = tid & 63;           // 0..63
  const int bkg   = (tid >> 6) * 8;     // 0,8,16,24

  union { uint4 u[2]; __bf16 h[16]; } aReg;
  union { uint4 u;    __bf16 h[8];  } bReg;

  // Issue A fetch for K-tile k0 into LDS buffer `buf` (async) or regs (sync).
  auto issueA = [&](int k0, int buf) {
    if constexpr (ASYNC_A) {
      const __bf16* gp = Ah + (long)(bm0 + arow) * lda + k0 + aks;
      async_ld_b128(gp,     &sA[buf][arow][aks]);
      async_ld_b128(gp + 8, &sA[buf][arow][aks + 8]);
    } else if constexpr (ABF16) {
      const __bf16* p = Ah + (long)(bm0 + arow) * lda + k0 + aks;
      aReg.u[0] = *(const uint4*)(p);
      aReg.u[1] = *(const uint4*)(p + 8);
    } else {
      const float* p = Af + (long)(bm0 + arow) * lda + k0 + aks;
#pragma unroll
      for (int v = 0; v < 4; ++v) {
        float4 f = *(const float4*)(p + v * 4);
        aReg.h[v * 4 + 0] = (__bf16)f.x;
        aReg.h[v * 4 + 1] = (__bf16)f.y;
        aReg.h[v * 4 + 2] = (__bf16)f.z;
        aReg.h[v * 4 + 3] = (__bf16)f.w;
      }
    }
  };
  auto gloadB = [&](int k0) {
#pragma unroll
    for (int i = 0; i < 8; ++i) {       // coalesced across lanes (consecutive n)
      float f = Bz[(long)(k0 + bkg + i) * ldb + bn0 + bncol];
      bReg.h[i] = (__bf16)f;
    }
  };
  auto lstore = [&](int buf) {
    if constexpr (!ASYNC_A) {
      *(uint4*)&sA[buf][arow][aks]     = aReg.u[0];
      *(uint4*)&sA[buf][arow][aks + 8] = aReg.u[1];
    }
    *(uint4*)&sB[buf][bncol][bkg] = bReg.u;
  };

  v8f acc[2][2];
#pragma unroll
  for (int i = 0; i < 2; ++i)
#pragma unroll
    for (int j = 0; j < 2; ++j)
#pragma unroll
      for (int r = 0; r < 8; ++r) acc[i][j][r] = 0.0f;

  const int KT = K / BK;
  issueA(0, 0); gloadB(0); lstore(0);
  if constexpr (ASYNC_A) wait_async0();
  __syncthreads();

  const int kb = (lane >> 4) << 3;      // 0 (lanes 0-15) or 8 (lanes 16-31)
  const int ml = lane & 15;

  for (int kt = 0; kt < KT; ++kt) {
    const int buf = kt & 1;
    const bool more = (kt + 1 < KT);
    if (more) { issueA((kt + 1) * BK, buf ^ 1); gloadB((kt + 1) * BK); }

    union { uint4 u[2]; v16bf v; } fa[2], fb[2];
#pragma unroll
    for (int i = 0; i < 2; ++i) {
      const __bf16* p = &sA[buf][wm * 32 + i * 16 + ml][kb];
      fa[i].u[0] = *(const uint4*)(p);        // K kb..kb+7
      fa[i].u[1] = *(const uint4*)(p + 16);   // K kb+16..kb+23
    }
#pragma unroll
    for (int j = 0; j < 2; ++j) {
      const __bf16* p = &sB[buf][wn * 32 + j * 16 + ml][kb];
      fb[j].u[0] = *(const uint4*)(p);
      fb[j].u[1] = *(const uint4*)(p + 16);
    }
#pragma unroll
    for (int i = 0; i < 2; ++i)
#pragma unroll
      for (int j = 0; j < 2; ++j)
        acc[i][j] = __builtin_amdgcn_wmma_f32_16x16x32_bf16(
            false, fa[i].v, false, fb[j].v, (short)0, acc[i][j], false, false);

    if (more) {
      lstore(buf ^ 1);
      if constexpr (ASYNC_A) wait_async0();   // async writes to buf^1 done
    }
    __syncthreads();
  }

  // C/D layout: VGPR r -> M = r (+8 for lanes 16-31), N = lane&15
#pragma unroll
  for (int i = 0; i < 2; ++i)
#pragma unroll
    for (int j = 0; j < 2; ++j) {
      const int row0 = bm0 + wm * 32 + i * 16 + ((lane >> 4) << 3);
      const int col  = bn0 + wn * 32 + j * 16 + ml;
#pragma unroll
      for (int r = 0; r < 8; ++r)
        Cz[(long)(row0 + r) * ldc + col] = acc[i][j][r];
    }
}

// ---------------------------------------------------------------------------
// Per-token: RMSNorm(q_low)->bf16; RMSNorm(latent[:512])->k_out+k_cache;
// interleaved RoPE(latent[512:576])->k_out+rope_cache.
// ---------------------------------------------------------------------------
__global__ __launch_bounds__(256)
void token_postprocess(const float* __restrict__ qkv,
                       const float* __restrict__ q_ln,
                       const float* __restrict__ kv_ln,
                       const float* __restrict__ cos_sin,
                       const int*   __restrict__ positions,
                       const int*   __restrict__ slots,
                       __bf16* __restrict__ q_norm,
                       float*  __restrict__ k_out,
                       float*  __restrict__ k_cache,
                       float*  __restrict__ rope_cache)
{
  const int t = blockIdx.x, tid = threadIdx.x;
  const float* row = qkv + (long)t * (QLORA + KVLORA + ROPE_D);
  const float* lat = row + QLORA;
  __shared__ float red[256];

  // --- RMSNorm over q_low (1536) ---
  float xq[6], s = 0.f;
#pragma unroll
  for (int e = 0; e < 6; ++e) { xq[e] = row[tid + e * 256]; s += xq[e] * xq[e]; }
  red[tid] = s; __syncthreads();
  for (int off = 128; off; off >>= 1) {
    if (tid < off) red[tid] += red[tid + off];
    __syncthreads();
  }
  const float qs = rsqrtf(red[0] * (1.0f / QLORA) + EPS_F);
  __syncthreads();
#pragma unroll
  for (int e = 0; e < 6; ++e) {
    const int i = tid + e * 256;
    q_norm[(long)t * QLORA + i] = (__bf16)(xq[e] * qs * q_ln[i]);
  }

  // --- RMSNorm over latent[:512] ---
  const float y0 = lat[tid], y1 = lat[tid + 256];
  red[tid] = y0 * y0 + y1 * y1; __syncthreads();
  for (int off = 128; off; off >>= 1) {
    if (tid < off) red[tid] += red[tid + off];
    __syncthreads();
  }
  const float ks = rsqrtf(red[0] * (1.0f / KVLORA) + EPS_F);
  const int slot = slots[t];
  const float kn0 = y0 * ks * kv_ln[tid];
  const float kn1 = y1 * ks * kv_ln[tid + 256];
  k_out[(long)t * 576 + tid]         = kn0;
  k_out[(long)t * 576 + tid + 256]   = kn1;
  k_cache[(long)slot * 512 + tid]       = kn0;
  k_cache[(long)slot * 512 + tid + 256] = kn1;

  // --- interleaved RoPE on latent[512:576] ---
  if (tid < 64) {
    const float* cs = cos_sin + (long)positions[t] * 64;  // [cos(32) | sin(32)]
    const float* x  = lat + 512;
    float o;
    if (tid < 32) o = x[2 * tid] * cs[tid] - x[2 * tid + 1] * cs[32 + tid];
    else { const int j = tid - 32; o = x[2 * j + 1] * cs[j] + x[2 * j] * cs[32 + j]; }
    k_out[(long)t * 576 + 512 + tid]  = o;
    rope_cache[(long)slot * 64 + tid] = o;
  }
}

// ---------------------------------------------------------------------------
// Per-token: q_nope f32->bf16 (for head GEMM) and RoPE(q_pe)->q_out[...,512:]
// ---------------------------------------------------------------------------
__global__ __launch_bounds__(256)
void q_postprocess(const float* __restrict__ qf,
                   const float* __restrict__ cos_sin,
                   const int*   __restrict__ positions,
                   __bf16* __restrict__ qnope,
                   float*  __restrict__ q_out)
{
  const int t = blockIdx.x, tid = threadIdx.x;
  const float* row = qf + (long)t * (NHEAD * QKD);
  __shared__ float cs[64];
  if (tid < 64) cs[tid] = cos_sin[(long)positions[t] * 64 + tid];
  __syncthreads();

#pragma unroll
  for (int e = 0; e < 8; ++e) {               // 16 heads * 128 nope = 2048
    const int idx = tid + e * 256;
    const int h = idx >> 7, j = idx & 127;
    qnope[(long)t * (NHEAD * NOPE_D) + idx] = (__bf16)row[h * QKD + j];
  }
#pragma unroll
  for (int e = 0; e < 4; ++e) {               // 16 heads * 64 pe = 1024
    const int idx = tid + e * 256;
    const int h = idx >> 6, i = idx & 63;
    const float* x = row + h * QKD + NOPE_D;
    float o;
    if (i < 32) o = x[2 * i] * cs[i] - x[2 * i + 1] * cs[32 + i];
    else { const int j = i - 32; o = x[2 * j + 1] * cs[j] + x[2 * j] * cs[32 + j]; }
    q_out[(long)t * (NHEAD * 576) + h * 576 + 512 + i] = o;
  }
}

// ---------------------------------------------------------------------------
extern "C" void kernel_launch(void* const* d_in, const int* in_sizes, int n_in,
                              void* d_out, int out_size, void* d_ws, size_t ws_size,
                              hipStream_t stream)
{
  const float* hidden        = (const float*)d_in[0];
  const int*   positions     = (const int*)  d_in[1];
  const float* w_qkv_a       = (const float*)d_in[2];
  const float* q_ln          = (const float*)d_in[3];
  const float* w_q_b         = (const float*)d_in[4];
  const float* kv_ln         = (const float*)d_in[5];
  const float* w_kc          = (const float*)d_in[6];
  const float* cos_sin       = (const float*)d_in[7];
  const float* k_cache_in    = (const float*)d_in[8];
  const float* rope_cache_in = (const float*)d_in[9];
  const int*   slots         = (const int*)  d_in[10];

  // d_out: q_out | k_out | k_cache | rope_cache (flat, f32)
  float* q_out      = (float*)d_out;
  float* k_out      = q_out + (size_t)NTOK * NHEAD * 576;
  float* k_cache    = k_out + (size_t)NTOK * 576;
  float* rope_cache = k_cache + (size_t)NSLOT * KVLORA;

  // workspace
  char* ws = (char*)d_ws;
  float*  qkv    = (float*)ws;  ws += (size_t)NTOK * (QLORA + KVLORA + ROPE_D) * sizeof(float);
  __bf16* q_norm = (__bf16*)ws; ws += (size_t)NTOK * QLORA * sizeof(__bf16);
  float*  qf     = (float*)ws;  ws += (size_t)NTOK * NHEAD * QKD * sizeof(float);
  __bf16* qnope  = (__bf16*)ws; ws += (size_t)NTOK * NHEAD * NOPE_D * sizeof(__bf16);

  // caches: base copy (zeros from input) then scatter-overwrite in token kernel
  (void)hipMemcpyAsync(k_cache, k_cache_in, (size_t)NSLOT * KVLORA * sizeof(float),
                       hipMemcpyDeviceToDevice, stream);
  (void)hipMemcpyAsync(rope_cache, rope_cache_in, (size_t)NSLOT * ROPE_D * sizeof(float),
                       hipMemcpyDeviceToDevice, stream);

  // GEMM1: qkv = hidden[4096,7168] @ w_qkv_a[7168,2112]
  dim3 g1(2112 / 64, NTOK / 128, 1);
  gemm_wmma_bf16<false><<<g1, 256, 0, stream>>>(
      hidden, w_qkv_a, qkv, NTOK, 2112, HID, HID, 2112, 2112, 0, 0, 0);

  token_postprocess<<<NTOK, 256, 0, stream>>>(
      qkv, q_ln, kv_ln, cos_sin, positions, slots, q_norm, k_out, k_cache, rope_cache);

  // GEMM2: qf = q_norm[4096,1536](bf16) @ w_q_b[1536,3072]
  dim3 g2(3072 / 64, NTOK / 128, 1);
  gemm_wmma_bf16<true><<<g2, 256, 0, stream>>>(
      q_norm, w_q_b, qf, NTOK, 3072, QLORA, QLORA, 3072, 3072, 0, 0, 0);

  q_postprocess<<<NTOK, 256, 0, stream>>>(qf, cos_sin, positions, qnope, q_out);

  // GEMM3 (batched over 16 heads): q_out[:,h,0:512] = qnope_h[4096,128] @ w_kc_h[128,512]
  dim3 g3(512 / 64, NTOK / 128, NHEAD);
  gemm_wmma_bf16<true><<<g3, 256, 0, stream>>>(
      qnope, w_kc, q_out, NTOK, 512, NOPE_D,
      NHEAD * NOPE_D, 512, NHEAD * 576,
      /*aBatch=*/NOPE_D, /*bBatch=*/(long)NOPE_D * 512, /*cBatch=*/576);
}
